// DendSeqNetSVHN3_45930380263778
// MI455X (gfx1250) — matrix operand order repaired and checked by reading, not compiled, not run on hardware
//
#include <hip/hip_runtime.h>
#include <hip/hip_bf16.h>

// ---------------------------------------------------------------------------
// DendSeqNetSVHN3: 100-step LIF spiking net.
// Phase 1: time-parallel WMMA GEMM  inj[cd,t,b,h] = x[t,b,c,:] . W_h[c,d,h,:] + b_h
//          128x64 block tile, 32x32 wave tile (2x2 WMMA), double-buffered LDS.
// Phase 2: time-sequential LIF scan, state-in-registers, one block per batch b.
// ---------------------------------------------------------------------------

typedef __fp16   v2hp __attribute__((ext_vector_type(2)));   // cvt_pkrtz result type
typedef __fp16   v4hp __attribute__((ext_vector_type(4)));
typedef _Float16 v4h  __attribute__((ext_vector_type(4)));
typedef _Float16 v8h  __attribute__((ext_vector_type(8)));
typedef _Float16 v16h __attribute__((ext_vector_type(16)));
typedef float    v8f  __attribute__((ext_vector_type(8)));

#define T_TOT 100
#define BB    256
#define CC    3
#define DD    3
#define CDN   9          // C*D
#define HH    200
#define KIN   1024       // FS*FS
#define OO    4
#define NOUTN 10
#define KOUT  50         // H / O

#define TILE_M 128
#define TILE_N 64
#define TILE_K 32
#define LDS_STR 40       // f16 elems/row: 80B rows -> 16B-aligned b128 frag loads

static __device__ __forceinline__ v4h pk4(float a, float b, float c, float d) {
    v2hp lo = __builtin_amdgcn_cvt_pkrtz(a, b);   // v_cvt_pk_rtz_f16_f32
    v2hp hi = __builtin_amdgcn_cvt_pkrtz(c, d);
    v4hp v  = __builtin_shufflevector(lo, hi, 0, 1, 2, 3);
    return __builtin_bit_cast(v4h, v);
}

static __device__ __forceinline__ v16h ldfrag(const _Float16* p) {
    v8h lo = *(const v8h*)p;
    v8h hi = *(const v8h*)(p + 16);   // A frag: K halves split by 16 elems
    return __builtin_shufflevector(lo, hi,
        0,1,2,3,4,5,6,7,8,9,10,11,12,13,14,15);
}

static __device__ __forceinline__ v16h ldfragB(const _Float16* p) {
    v8h lo = *(const v8h*)p;
    v8h hi = *(const v8h*)(p + 16);   // B frag: K halves split by 16 elems
    return __builtin_shufflevector(lo, hi,
        0,1,2,3,4,5,6,7,8,9,10,11,12,13,14,15);
}

// ---------------------------------------------------------------------------
// Phase 1. grid: (8 = 2 mtiles * 4 ntiles, 9 = cd, tch), block 256 (8 waves)
// ---------------------------------------------------------------------------
__global__ __launch_bounds__(256)
void dendnet_phase1_wmma(const float* __restrict__ x,
                         const float* __restrict__ Wh,
                         const float* __restrict__ bh,
                         float* __restrict__ inj,
                         int t0, int tch)
{
    __shared__ __align__(16) _Float16 sA[2][TILE_M * LDS_STR];
    __shared__ __align__(16) _Float16 sB[2][TILE_N * LDS_STR];

    const int tid  = threadIdx.x;
    const int lane = tid & 31;
    const int wave = tid >> 5;              // 0..7
    const int mt   = blockIdx.x & 1;        // 2 b-tiles of 128
    const int nt   = blockIdx.x >> 1;       // 4 h-tiles of 64
    const int cd   = blockIdx.y;            // 0..8
    const int tl   = blockIdx.z;            // chunk-local t
    const int t    = t0 + tl;
    const int c    = cd / DD;

    const int b0 = mt * TILE_M;
    const int n0 = nt * TILE_N;

    // wave -> 32x32 sub-tile: 4 M-positions x 2 N-positions
    const int wm = (wave & 3) * 32;
    const int wn = (wave >> 2) * 32;

    v8f acc00 = {}, acc01 = {}, acc10 = {}, acc11 = {};

    const float* xbase = x  + ((size_t)t * BB * CC + c) * KIN;  // + b*CC*KIN + i
    const float* wbase = Wh + (size_t)cd * HH * KIN;            // + h*KIN + i

    // ---- staging geometry ------------------------------------------------
    // A: 128x32 f32 = 1024 float4 slots, 4/thread; B: 64x32 = 512 slots, 2/thread.
    // slot = tid + i*256; since 256 % 8 == 0: col fixed, row steps by 32.
    const int rr  = tid >> 3;                 // 0..31
    const int cS  = (tid & 7) * 4;            // 0,4,..,28

    const float* pa[4];
    #pragma unroll
    for (int i = 0; i < 4; ++i)
        pa[i] = xbase + (size_t)(b0 + rr + 32 * i) * CC * KIN + cS;

    const float* pb[2];
    float mb[2];
    #pragma unroll
    for (int i = 0; i < 2; ++i) {
        int hb = n0 + rr + 32 * i;
        int hc = (hb < HH) ? hb : (HH - 1);   // clamp: branch-free OOB
        mb[i] = (hb < HH) ? 1.0f : 0.0f;
        pb[i] = wbase + (size_t)hc * KIN + cS;
    }

    // ---- per-thread fragment offsets (f16 element index within a buffer) --
    // A frag 16x32: lanes 0-15 K={0..7,16..23}; lanes 16-31 K={8..15,24..31}
    const int aoff0 = (wm + (lane & 15)) * LDS_STR + (lane >> 4) * 8;
    const int aoff1 = aoff0 + 16 * LDS_STR;
    // B frag 32x16: lanes 0-15 K={0..7,16..23}; lanes 16-31 K={8..15,24..31}
    const int boff0 = (wn + (lane & 15)) * LDS_STR + (lane >> 4) * 8;
    const int boff1 = boff0 + 16 * LDS_STR;

    float4 av[4], bv[2];

    auto stage = [&](int bufi) {
        #pragma unroll
        for (int i = 0; i < 4; ++i)
            *(v4h*)&sA[bufi][(rr + 32 * i) * LDS_STR + cS] =
                pk4(av[i].x, av[i].y, av[i].z, av[i].w);
        #pragma unroll
        for (int i = 0; i < 2; ++i)
            *(v4h*)&sB[bufi][(rr + 32 * i) * LDS_STR + cS] =
                pk4(bv[i].x * mb[i], bv[i].y * mb[i],
                    bv[i].z * mb[i], bv[i].w * mb[i]);
    };

    auto compute = [&](int bufi) {
        v16h af0 = ldfrag(&sA[bufi][aoff0]);
        v16h af1 = ldfrag(&sA[bufi][aoff1]);
        v16h bf0 = ldfragB(&sB[bufi][boff0]);
        v16h bf1 = ldfragB(&sB[bufi][boff1]);
        acc00 = __builtin_amdgcn_wmma_f32_16x16x32_f16(
            false, af0, false, bf0, (short)0, acc00, false, false);
        acc01 = __builtin_amdgcn_wmma_f32_16x16x32_f16(
            false, af0, false, bf1, (short)0, acc01, false, false);
        acc10 = __builtin_amdgcn_wmma_f32_16x16x32_f16(
            false, af1, false, bf0, (short)0, acc10, false, false);
        acc11 = __builtin_amdgcn_wmma_f32_16x16x32_f16(
            false, af1, false, bf1, (short)0, acc11, false, false);
    };

    // ---- prologue: stage K-chunk 0 into buffer 0 ----
    #pragma unroll
    for (int i = 0; i < 4; ++i) av[i] = *(const float4*)pa[i];
    #pragma unroll
    for (int i = 0; i < 2; ++i) bv[i] = *(const float4*)pb[i];
    stage(0);
    __syncthreads();

    // ---- main loop: load chunk k, compute chunk k-1, flip ----
    int buf = 0;
    for (int kc = TILE_K; kc < KIN; kc += TILE_K) {
        #pragma unroll
        for (int i = 0; i < 4; ++i) av[i] = *(const float4*)(pa[i] + kc);
        #pragma unroll
        for (int i = 0; i < 2; ++i) bv[i] = *(const float4*)(pb[i] + kc);
        compute(buf);              // consumes sX[buf]; global loads in flight
        stage(buf ^ 1);            // waits loads, fills other buffer
        __syncthreads();
        buf ^= 1;
    }
    compute(buf);                  // last K-chunk

    // ---- epilogue: C/D layout -> inj + bias ----
    // 16x16 f32 D: lane n = lane&15; VGPR r holds M = r + 8*(lane>=16)
    const int nlane = lane & 15;
    const int mhalf = (lane >> 4) * 8;
    const v8f* accs[4] = { &acc00, &acc01, &acc10, &acc11 };
    #pragma unroll
    for (int s = 0; s < 2; ++s) {          // A strip (rows +0 / +16)
        #pragma unroll
        for (int tn = 0; tn < 2; ++tn) {   // B tile (cols +0 / +16)
            const v8f& a = *accs[s * 2 + tn];
            int h = n0 + wn + tn * 16 + nlane;
            if (h < HH) {
                float bias = bh[cd * HH + h];
                #pragma unroll
                for (int r = 0; r < 8; ++r) {
                    int b = b0 + wm + s * 16 + mhalf + r;
                    size_t base = (((size_t)cd * tch + tl) * BB + b) * HH;
                    inj[base + h] = a[r] + bias;
                }
            }
        }
    }
}

// ---------------------------------------------------------------------------
// Phase 2: sequential LIF scan over one T-chunk. grid = (256 = B), block 256.
// thread h < 200 carries all 9 (c,d) LIF states in registers.
// ---------------------------------------------------------------------------
__global__ __launch_bounds__(256)
void dendnet_phase2_scan(const float* __restrict__ inj,
                         const float* __restrict__ Wo,
                         const float* __restrict__ bo,
                         float* __restrict__ out,
                         float* __restrict__ state,
                         int t0, int tch)
{
    __shared__ float zsh[HH];
    __shared__ float vstage[OO * NOUTN];
    __shared__ float sWo[OO * NOUTN * KOUT];   // 2000 floats
    __shared__ float sbo[OO * NOUTN];

    const int tid = threadIdx.x;
    const int b   = blockIdx.x;

    for (int i = tid; i < OO * NOUTN * KOUT; i += 256) sWo[i] = Wo[i];
    if (tid < OO * NOUTN) sbo[tid] = bo[tid];

    float* vh_s = state;
    float* ih_s = state + (size_t)CDN * BB * HH;
    float* vo_s = state + (size_t)2 * CDN * BB * HH;
    float* io_s = vo_s + (size_t)OO * BB * NOUTN;

    const int h = tid;
    float vh[CDN], ih[CDN];
    if (h < HH) {
        #pragma unroll
        for (int cd = 0; cd < CDN; ++cd) {
            if (t0 == 0) { vh[cd] = 0.0f; ih[cd] = 0.0f; }
            else {
                vh[cd] = vh_s[((size_t)cd * BB + b) * HH + h];
                ih[cd] = ih_s[((size_t)cd * BB + b) * HH + h];
            }
        }
    }
    const int o = tid / NOUTN;
    const int n = tid - o * NOUTN;
    float vo = 0.0f, io = 0.0f;
    if (tid < OO * NOUTN && t0 != 0) {
        vo = vo_s[((size_t)o * BB + b) * NOUTN + n];
        io = io_s[((size_t)o * BB + b) * NOUTN + n];
    }
    __syncthreads();

    for (int tl = 0; tl < tch; ++tl) {
        if (h < HH) {
            float zsum = 0.0f;
            #pragma unroll
            for (int cd = 0; cd < CDN; ++cd) {
                float injv = inj[(((size_t)cd * tch + tl) * BB + b) * HH + h];
                float vdec = 0.9f * vh[cd] + 0.1f * ih[cd];   // v + dt*tmi*(-v+i)
                float idec = 0.8f * ih[cd];                   // i*(1-dt*tsi)
                float sp   = (vdec - 1.0f) > 0.0f ? 1.0f : 0.0f;
                vh[cd] = vdec - sp * vdec;                    // (1-z)*vdec
                ih[cd] = idec + injv;
                zsum  += sp;
            }
            zsh[h] = zsum;
        }
        __syncthreads();
        if (tid < OO * NOUTN) {
            float acc = 0.0f;
            #pragma unroll
            for (int k = 0; k < KOUT; ++k)
                acc += zsh[o * KOUT + k] * sWo[(o * NOUTN + n) * KOUT + k];
            float von = 0.9f * vo + 0.1f * io;
            io = 0.8f * io + acc + sbo[tid];
            vo = von;
            vstage[tid] = von;
        }
        __syncthreads();
        if (tid < NOUTN) {
            int t = t0 + tl;
            out[((size_t)t * BB + b) * NOUTN + tid] =
                vstage[tid] + vstage[NOUTN + tid] +
                vstage[2 * NOUTN + tid] + vstage[3 * NOUTN + tid];
        }
        // barrier at top of next iteration fences vstage reads vs next writes;
        // zsh reads are fenced by the second barrier.
    }

    // persist state for next chunk
    if (h < HH) {
        #pragma unroll
        for (int cd = 0; cd < CDN; ++cd) {
            vh_s[((size_t)cd * BB + b) * HH + h] = vh[cd];
            ih_s[((size_t)cd * BB + b) * HH + h] = ih[cd];
        }
    }
    if (tid < OO * NOUTN) {
        vo_s[((size_t)o * BB + b) * NOUTN + n] = vo;
        io_s[((size_t)o * BB + b) * NOUTN + n] = io;
    }
}

// ---------------------------------------------------------------------------
extern "C" void kernel_launch(void* const* d_in, const int* in_sizes, int n_in,
                              void* d_out, int out_size, void* d_ws, size_t ws_size,
                              hipStream_t stream) {
    const float* x  = (const float*)d_in[0];   // (T,B,C,FS,FS)
    const float* Wh = (const float*)d_in[1];   // (C,D,H,1024)
    const float* bh = (const float*)d_in[2];   // (C,D,H)
    const float* Wo = (const float*)d_in[3];   // (O,10,50)
    const float* bo = (const float*)d_in[4];   // (O,10)
    float* out = (float*)d_out;                // (T,B,10)

    const size_t stateFloats =
        (size_t)2 * CDN * BB * HH + (size_t)2 * OO * BB * NOUTN;

    // largest divisor of T fitting ws: inj chunk + persistent state
    const int cands[8] = {50, 25, 20, 10, 5, 4, 2, 1};
    int tch = 1;
    for (int i = 0; i < 8; ++i) {
        size_t need = ((size_t)CDN * cands[i] * BB * HH + stateFloats) * sizeof(float);
        if (need <= ws_size) { tch = cands[i]; break; }
    }

    float* inj   = (float*)d_ws;
    float* state = inj + (size_t)CDN * tch * BB * HH;

    dim3 g1(8, CDN, tch);
    for (int t0 = 0; t0 < T_TOT; t0 += tch) {
        dendnet_phase1_wmma<<<g1, 256, 0, stream>>>(x, Wh, bh, inj, t0, tch);
        dendnet_phase2_scan<<<dim3(BB), 256, 0, stream>>>(inj, Wo, bo, out, state, t0, tch);
    }
}
